// Attn_3375844295368
// MI455X (gfx1250) — matrix operand order repaired and checked by reading, not compiled, not run on hardware
//
#include <hip/hip_runtime.h>
#include <hip/hip_bf16.h>

#define DIM   512
#define QK    128
#define BATCH 4
#define NSEQ  4096

typedef __attribute__((ext_vector_type(16))) _Float16 v16h;
typedef __attribute__((ext_vector_type(8)))  _Float16 v8h;
typedef __attribute__((ext_vector_type(4)))  _Float16 v4h;
typedef __attribute__((ext_vector_type(8)))  float    v8f;

// ---------------------------------------------------------------------------
// Kernel 0: convert w_qk (128x512 f32) -> f16, read once per block later.
// ---------------------------------------------------------------------------
__global__ __launch_bounds__(256) void cvt_w_kernel(
    const float* __restrict__ w, _Float16* __restrict__ wh)
{
    const int i = (blockIdx.x * 256 + threadIdx.x) * 4;   // 64 blocks cover 65536
    float4 v = *(const float4*)(w + i);
    v4h h;
    h.x = (_Float16)v.x; h.y = (_Float16)v.y;
    h.z = (_Float16)v.z; h.w = (_Float16)v.w;
    *(v4h*)(wh + i) = h;
}

// ---------------------------------------------------------------------------
// Kernel 1: fused LayerNorm + WMMA projection (512->128) + SiLU + OffsetScale.
// One block (256 thr = 8 waves) per 16-row strip. Each wave owns one 16-wide
// output-channel tile; K=512 -> 16 v_wmma_f32_16x16x32_f16 steps.
// w_qk(f16) is read exactly once per block: 128 KB * 1024 blocks = 128 MB L2.
// ---------------------------------------------------------------------------
__global__ __launch_bounds__(256) void ln_proj_kernel(
    const float* __restrict__ x, const float* __restrict__ ln_w,
    const float* __restrict__ ln_b, const _Float16* __restrict__ w_h,
    const float* __restrict__ b_qk, const float* __restrict__ gamma,
    const float* __restrict__ beta,
    _Float16* __restrict__ qo, _Float16* __restrict__ ko)
{
    __shared__ _Float16 nrm[16][520];     // padded: 520*2B stride kills bank conflicts
    __shared__ float red1[16][16];
    __shared__ float red2[16][16];
    __shared__ float mrs[16][2];

    const int tid = threadIdx.x;
    const int r   = tid >> 4;             // row 0..15 of strip
    const int c   = tid & 15;             // 32-float chunk within the row
    const size_t row0 = (size_t)blockIdx.x * 16;

    // --- LayerNorm statistics (16 threads per row, 32 elems each) ---
    const float* xr = x + (row0 + r) * DIM + c * 32;
    float4 xv[8];
    float s = 0.f, ss = 0.f;
    #pragma unroll
    for (int i = 0; i < 8; ++i) {
        xv[i] = ((const float4*)xr)[i];
        s  += xv[i].x + xv[i].y + xv[i].z + xv[i].w;
        ss += xv[i].x * xv[i].x + xv[i].y * xv[i].y
            + xv[i].z * xv[i].z + xv[i].w * xv[i].w;
    }
    red1[r][c] = s; red2[r][c] = ss;
    __syncthreads();
    if (c == 0) {
        float t1 = 0.f, t2 = 0.f;
        #pragma unroll
        for (int i = 0; i < 16; ++i) { t1 += red1[r][i]; t2 += red2[r][i]; }
        const float mean = t1 * (1.0f / DIM);
        const float var  = t2 * (1.0f / DIM) - mean * mean;
        mrs[r][0] = mean;
        mrs[r][1] = rsqrtf(var + 1e-5f);
    }
    __syncthreads();
    const float mean = mrs[r][0], rstd = mrs[r][1];
    #pragma unroll
    for (int i = 0; i < 8; ++i) {
        const int j = c * 32 + i * 4;
        float4 wv = *(const float4*)(ln_w + j);
        float4 bv = *(const float4*)(ln_b + j);
        nrm[r][j + 0] = (_Float16)((xv[i].x - mean) * rstd * wv.x + bv.x);
        nrm[r][j + 1] = (_Float16)((xv[i].y - mean) * rstd * wv.y + bv.y);
        nrm[r][j + 2] = (_Float16)((xv[i].z - mean) * rstd * wv.z + bv.z);
        nrm[r][j + 3] = (_Float16)((xv[i].w - mean) * rstd * wv.w + bv.w);
    }
    __syncthreads();

    // --- WMMA projection: C(16x128) = nrm(16x512) * w^T(512x128) ---
    const int wave = tid >> 5, lane = tid & 31;
    const int lr = lane & 15;             // A row / B col within tile
    const int hi = lane >> 4;

    // B: B[d][e] = w_qk[e][d] -> lane (col e) reads w_h row contiguously
    const _Float16* wrow = w_h + (size_t)(wave * 16 + lr) * DIM;

    v8f acc = {};
    #pragma unroll
    for (int s2 = 0; s2 < 16; ++s2) {
        v8h lo  = *(const v8h*)&nrm[lr][32 * s2 + 8 * hi];        // K 8hi..+7
        v8h hi8 = *(const v8h*)&nrm[lr][32 * s2 + 8 * hi + 16];   // K 8hi+16..+7
        v16h afrag = __builtin_shufflevector(lo, hi8, 0, 1, 2, 3, 4, 5, 6, 7,
                                                      8, 9, 10, 11, 12, 13, 14, 15);
        v8h blo = *(const v8h*)(wrow + 32 * s2 + 8 * hi);
        v8h bhi = *(const v8h*)(wrow + 32 * s2 + 8 * hi + 16);
        v16h bfrag = __builtin_shufflevector(blo, bhi, 0, 1, 2, 3, 4, 5, 6, 7,
                                                      8, 9, 10, 11, 12, 13, 14, 15);
        acc = __builtin_amdgcn_wmma_f32_16x16x32_f16(
                  false, afrag, false, bfrag, (short)0, acc, false, false);
    }

    // --- epilogue: bias + SiLU + offset-scale -> q,k f16 ---
    const int e = wave * 16 + lr;
    const float bb  = b_qk[e];
    const float g0  = gamma[e],      g1  = gamma[QK + e];
    const float be0 = beta[e],       be1 = beta[QK + e];
    #pragma unroll
    for (int rr = 0; rr < 8; ++rr) {
        const int orow = rr + 8 * hi;                 // C layout row
        const float v  = acc[rr] + bb;
        const float sv = v / (1.0f + __expf(-v));     // SiLU
        qo[(row0 + orow) * QK + e] = (_Float16)(sv * g0 + be0);
        ko[(row0 + orow) * QK + e] = (_Float16)(sv * g1 + be1);
    }
}

// ---------------------------------------------------------------------------
// Kernel 2: attention strip. One block (8 waves) owns 16 query rows.
// sim tile via v_wmma_f32_16x16x32_f16 (K=128 -> 4 WMMAs/tile), relu^2,
// 16x4096 f32 strip resident in LDS (256KB), row sums via ds_add_f32,
// single normalized streaming store (output-bandwidth bound: 256 MB once).
// ---------------------------------------------------------------------------
__global__ __launch_bounds__(256) void attn_strip_kernel(
    const _Float16* __restrict__ q, const _Float16* __restrict__ k,
    float* __restrict__ out)
{
    extern __shared__ float smem[];          // 16*NSEQ sim + 16 rowsum
    float* simbuf = smem;
    float* rowsum = smem + 16 * NSEQ;

    const int tid  = threadIdx.x;
    const int wave = tid >> 5;
    const int lane = tid & 31;
    const int b     = blockIdx.x >> 8;       // 256 strips per batch
    const int strip = blockIdx.x & 255;
    const int i0    = strip << 4;

    if (tid < 16) rowsum[tid] = 1e-6f;       // eps of the reference
    __syncthreads();

    const int lr = lane & 15;                // row (A) / col (B) within tile
    const int hi = lane >> 4;                // half-wave select
    const int kb = hi * 8;                   // A-frag K base & C row offset

    // --- A fragments: 16x32 f16 per ISA layout, 4 steps cover K=128 ---
    const _Float16* qrow = q + (size_t)(b * NSEQ + i0 + lr) * QK;
    v16h a[4];
    #pragma unroll
    for (int s2 = 0; s2 < 4; ++s2) {
        const int base = 32 * s2 + kb;
        v8h lo  = *(const v8h*)(qrow + base);        // K = kb .. kb+7
        v8h hi8 = *(const v8h*)(qrow + base + 16);   // K = kb+16 .. kb+23
        a[s2] = __builtin_shufflevector(lo, hi8, 0, 1, 2, 3, 4, 5, 6, 7,
                                                 8, 9, 10, 11, 12, 13, 14, 15);
    }

    float ssum[8] = {0.f, 0.f, 0.f, 0.f, 0.f, 0.f, 0.f, 0.f};
    const int kbase16 = hi * 16;             // unused; B-frag K base is kb

    // Each wave sweeps 32 consecutive 16-wide key tiles (8 waves * 32 = 256)
    const int jt0 = wave * 32;
    for (int jt = jt0; jt < jt0 + 32; ++jt) {
        const int jb = jt << 4;
        const _Float16* krow = k + (size_t)(b * NSEQ + jb + lr) * QK;
        v8f c = {};
        #pragma unroll
        for (int s2 = 0; s2 < 4; ++s2) {
            v8h blo = *(const v8h*)(krow + 32 * s2 + kb);
            v8h bhi = *(const v8h*)(krow + 32 * s2 + kb + 16);
            v16h bfrag = __builtin_shufflevector(blo, bhi, 0, 1, 2, 3, 4, 5, 6, 7,
                                                          8, 9, 10, 11, 12, 13, 14, 15);
            c = __builtin_amdgcn_wmma_f32_16x16x32_f16(
                    false, a[s2], false, bfrag, (short)0, c, false, false);
        }
        #pragma unroll
        for (int r = 0; r < 8; ++r) {
            float v = c[r] * (1.0f / 64.0f);          // / sqrt(N)
            v = fmaxf(v, 0.0f);
            v = v * v;                                 // relu^2
            simbuf[(size_t)(r + kb) * NSEQ + jb + lr] = v;
            ssum[r] += v;
        }
    }
    #pragma unroll
    for (int r = 0; r < 8; ++r) atomicAdd(&rowsum[r + kb], ssum[r]);
    __syncthreads();

    if (tid < 16) rowsum[tid] = 1.0f / rowsum[tid];   // sums already hold +eps
    __syncthreads();

    // Strip rows are contiguous in the row-major [B,N,N] output.
    float* obase = out + (size_t)(b * NSEQ + i0) * NSEQ;
    for (int idx = tid * 4; idx < 16 * NSEQ; idx += 256 * 4) {
        float4 v = *(float4*)&simbuf[idx];
        const float inv = rowsum[idx >> 12];          // idx/4096 = row
        v.x *= inv; v.y *= inv; v.z *= inv; v.w *= inv;
        *(float4*)&obase[idx] = v;
    }
}

// ---------------------------------------------------------------------------
extern "C" void kernel_launch(void* const* d_in, const int* in_sizes, int n_in,
                              void* d_out, int out_size, void* d_ws, size_t ws_size,
                              hipStream_t stream)
{
    const float* x     = (const float*)d_in[0];
    const float* ln_w  = (const float*)d_in[1];
    const float* ln_b  = (const float*)d_in[2];
    const float* w_qk  = (const float*)d_in[3];
    const float* b_qk  = (const float*)d_in[4];
    const float* gamma = (const float*)d_in[5];
    const float* beta  = (const float*)d_in[6];

    _Float16* qws = (_Float16*)d_ws;
    _Float16* kws = qws + (size_t)BATCH * NSEQ * QK;
    _Float16* whs = kws + (size_t)BATCH * NSEQ * QK;   // 128 KB f16 weights

    cvt_w_kernel<<<(QK * DIM) / (256 * 4), 256, 0, stream>>>(w_qk, whs);

    ln_proj_kernel<<<(BATCH * NSEQ) / 16, 256, 0, stream>>>(
        x, ln_w, ln_b, whs, b_qk, gamma, beta, qws, kws);

    const size_t smem = (16 * (size_t)NSEQ + 16) * sizeof(float);   // ~256 KB
    (void)hipFuncSetAttribute((const void*)attn_strip_kernel,
                              hipFuncAttributeMaxDynamicSharedMemorySize,
                              (int)smem);
    attn_strip_kernel<<<BATCH * 256, 256, smem, stream>>>(
        qws, kws, (float*)d_out);
}
